// CrossAttention3D_16415365005619
// MI455X (gfx1250) — matrix operand order repaired and checked by reading, not compiled, not run on hardware
//
#include <hip/hip_runtime.h>
#include <hip/hip_bf16.h>

// ---------------------------------------------------------------------------
// 3D shifted-window cross attention for MI455X (gfx1250, wave32, WMMA).
//   C=192, NH=8, DH=24 (padded to 32 for one f16 WMMA K-step)
//   window 6x8x8 -> N=384 tokens, nW=256 windows, shift (3,4,4)
// ---------------------------------------------------------------------------

#define C_DIM   192
#define NH      8
#define DH      24
#define DHP     32          // padded head dim (one 16x16x32 K-step)
#define NTOK    384         // tokens per window
#define NWIN    256
#define DD      24
#define HH      64
#define WW_     64
#define DHW     (DD*HH*WW_) // 98304
#define SCALE   0.20412414523193154f   // 1/sqrt(24)

typedef __attribute__((ext_vector_type(16))) _Float16 v16h;
typedef __attribute__((ext_vector_type(8)))  float    v8f;

union Frag { v16h v; unsigned u[8]; };

#define CFENCE() asm volatile("" ::: "memory")

__device__ __forceinline__ v8f wmma_f16(v16h a, v16h b, v8f c) {
    // D = A(16x32) * B(32x16) + C ; emits v_wmma_f32_16x16x32_f16
    return __builtin_amdgcn_wmma_f32_16x16x32_f16(
        /*neg_a=*/false, a, /*neg_b=*/false, b,
        /*c_mod=*/(short)0, c, /*reuse_a=*/false, /*reuse_b=*/false);
}

// A 16x32 f16 fragment from LDS, layout [m][k] with given f16 stride.
// lanes 0-15: m=lane,   K in {0..7, 16..23}
// lanes16-31: m=lane-16,K in {8..15, 24..31}   (2 f16 per dword)
__device__ __forceinline__ v16h load_frag_A(const _Float16* rowBase, int hi) {
    const unsigned* p = (const unsigned*)rowBase;
    Frag f; int db = hi * 4;
#pragma unroll
    for (int j = 0; j < 4; ++j) { f.u[j] = p[db + j]; f.u[4 + j] = p[db + 8 + j]; }
    return f.v;
}

// B 32x16 f16 fragment from LDS, layout [n][k] with rowBase at [n][k0].
// lanes 0-15: n=lane,   K in {0..7, 16..23}
// lanes16-31: n=lane-16,K in {8..15, 24..31}
__device__ __forceinline__ v16h load_frag_B(const _Float16* rowBase) {
    const unsigned* p = (const unsigned*)rowBase;
    int hi = (threadIdx.x >> 4) & 1;
    Frag f; int db = hi * 4;
#pragma unroll
    for (int j = 0; j < 4; ++j) { f.u[j] = p[db + j]; f.u[4 + j] = p[db + 8 + j]; }
    return f.v;
}

// ----- width-16 butterfly reductions via DPP (no LDS, no dscnt waits) ------
// steps: quad_perm(1,0,3,2)=0xB1, quad_perm(2,3,0,1)=0x4E,
//        row_half_mirror=0x141, row_mirror=0x140
template <int CTRL>
__device__ __forceinline__ float dpp_mov_f32(float x) {
    int xi = __builtin_bit_cast(int, x);
    int r  = __builtin_amdgcn_update_dpp(0, xi, CTRL, 0xf, 0xf, true);
    return __builtin_bit_cast(float, r);
}
__device__ __forceinline__ float redmax16(float x) {
    x = fmaxf(x, dpp_mov_f32<0xB1>(x));
    x = fmaxf(x, dpp_mov_f32<0x4E>(x));
    x = fmaxf(x, dpp_mov_f32<0x141>(x));
    x = fmaxf(x, dpp_mov_f32<0x140>(x));
    return x;
}
__device__ __forceinline__ float redsum16(float x) {
    x += dpp_mov_f32<0xB1>(x);
    x += dpp_mov_f32<0x4E>(x);
    x += dpp_mov_f32<0x141>(x);
    x += dpp_mov_f32<0x140>(x);
    return x;
}

// Swin shift-mask region id, built on the *unshifted* grid (matches reference)
__device__ __forceinline__ int region3(int x, int L, int Wn, int S) {
    return (x < L - Wn) ? 0 : ((x < L - S) ? 1 : 2);
}
__device__ __forceinline__ int group_id(int win, int t) {
    int wd = win >> 6, wh = (win >> 3) & 7, ww = win & 7;
    int td = t  >> 6, th = (t  >> 3) & 7, tw = t  & 7;
    return region3(wd * 6 + td, DD, 6, 3) * 9 +
           region3(wh * 8 + th, HH, 8, 4) * 3 +
           region3(ww * 8 + tw, WW_, 8, 4);
}

// ---------------------------------------------------------------------------
// Stage 1: cyclic shift + window partition + transpose + LayerNorm
//   in : (C, D, H, W) f32       out: (win, c, token) f16
//   grid (NWIN, 3), block 384 (one thread per token)
// ---------------------------------------------------------------------------
__global__ void ln_window_kernel(const float* __restrict__ qm,
                                 const float* __restrict__ km,
                                 const float* __restrict__ vm,
                                 const float* __restrict__ wq,
                                 const float* __restrict__ bq,
                                 const float* __restrict__ wkv,
                                 const float* __restrict__ bkv,
                                 _Float16* __restrict__ qn,
                                 _Float16* __restrict__ kn,
                                 _Float16* __restrict__ vn) {
    extern __shared__ char smem[];
    _Float16* xs = (_Float16*)smem;           // [192][384] = 147456 B

    int win = blockIdx.x, which = blockIdx.y;
    const float* in = (which == 0) ? qm : (which == 1) ? km : vm;
    const float* wgt = (which == 0) ? wq : wkv;
    const float* bia = (which == 0) ? bq : bkv;
    _Float16* out = ((which == 0) ? qn : (which == 1) ? kn : vn) +
                    (size_t)win * C_DIM * NTOK;

    int t = threadIdx.x;                       // 0..383
    int wd = win >> 6, wh = (win >> 3) & 7, ww = win & 7;
    int td = t  >> 6, th = (t  >> 3) & 7, tw = t  & 7;
    int d = (wd * 6 + td + 3) % DD;            // roll by -shift == +S source
    int h = (wh * 8 + th + 4) & 63;
    int w = (ww * 8 + tw + 4) & 63;
    size_t sp = (size_t)d * (HH * WW_) + h * WW_ + w;

    float sum = 0.f, sum2 = 0.f;
#pragma unroll 4
    for (int c = 0; c < C_DIM; ++c) {
        float x = in[(size_t)c * DHW + sp];
        sum += x; sum2 += x * x;
        xs[c * NTOK + t] = (_Float16)x;        // each thread owns column t
    }
    float mean = sum * (1.0f / C_DIM);
    float var  = sum2 * (1.0f / C_DIM) - mean * mean;
    float inv  = rsqrtf(var + 1e-5f);
#pragma unroll 4
    for (int c = 0; c < C_DIM; ++c) {
        float x = (float)xs[c * NTOK + t];
        out[(size_t)c * NTOK + t] = (_Float16)((x - mean) * inv * wgt[c] + bia[c]);
    }
}

// ---------------------------------------------------------------------------
// Stage 2: flash attention per (window, head) using f16 WMMA, online softmax
//   grid (NWIN, NH), block 256 (8 waves, 3 q-tiles of 16 each)
//   LDS: Qs[384][34] + Ks[384][34] + Vs[32][386] + Ps[8][16][34] = 85632 B
// ---------------------------------------------------------------------------
#define QS_STR 34
#define VS_STR 386
__global__ void attn_kernel(const _Float16* __restrict__ qn,
                            const _Float16* __restrict__ kn,
                            const _Float16* __restrict__ vn,
                            _Float16* __restrict__ attnout) {
    extern __shared__ char smem[];
    _Float16* Qs = (_Float16*)smem;                       // [384][34]
    _Float16* Ks = Qs + NTOK * QS_STR;                    // [384][34]
    _Float16* Vs = Ks + NTOK * QS_STR;                    // [32][386]
    _Float16* Ps = Vs + DHP * VS_STR;                     // [8][16][34]

    int win = blockIdx.x, head = blockIdx.y;
    int tid = threadIdx.x;
    const _Float16* qg = qn + ((size_t)win * C_DIM + head * DH) * NTOK;
    const _Float16* kg = kn + ((size_t)win * C_DIM + head * DH) * NTOK;
    const _Float16* vg = vn + ((size_t)win * C_DIM + head * DH) * NTOK;

    // stage + transpose (pad d 24..31 with zeros)
    for (int i = tid; i < NTOK * DHP; i += 256) {
        int t = i % NTOK, d = i / NTOK;
        _Float16 q = (_Float16)0.f, k = (_Float16)0.f, v = (_Float16)0.f;
        if (d < DH) {
            q = qg[(size_t)d * NTOK + t];
            k = kg[(size_t)d * NTOK + t];
            v = vg[(size_t)d * NTOK + t];
        }
        Qs[t * QS_STR + d] = q;
        Ks[t * QS_STR + d] = k;
        Vs[d * VS_STR + t] = v;
    }
    __syncthreads();

    int wave = tid >> 5, lane = tid & 31;
    int l15 = lane & 15, hi = lane >> 4;
    _Float16* Pw = Ps + wave * 16 * QS_STR;

    for (int qt = wave; qt < NTOK / 16; qt += 8) {
        int qbase = qt * 16;
        v16h qa = load_frag_A(Qs + (qbase + l15) * QS_STR, hi);

        int gq[8];
#pragma unroll
        for (int r = 0; r < 8; ++r) gq[r] = group_id(win, qbase + r + 8 * hi);

        float m[8], ls[8];
        v8f acc0 = {}, acc1 = {};
#pragma unroll
        for (int r = 0; r < 8; ++r) { m[r] = -3.0e38f; ls[r] = 0.f; }

        for (int kb = 0; kb < NTOK; kb += 32) {
            v16h bk0 = load_frag_B(Ks + (kb + l15) * QS_STR);
            v16h bk1 = load_frag_B(Ks + (kb + 16 + l15) * QS_STR);
            v8f z = {};
            v8f s0 = wmma_f16(qa, bk0, z);
            v8f s1 = wmma_f16(qa, bk1, z);

            int gk0 = group_id(win, kb + l15);
            int gk1 = group_id(win, kb + 16 + l15);

            float p0[8], p1[8];
            CFENCE();
#pragma unroll
            for (int r = 0; r < 8; ++r) {
                float a = s0[r] * SCALE + ((gk0 == gq[r]) ? 0.f : -1.0e30f);
                float b = s1[r] * SCALE + ((gk1 == gq[r]) ? 0.f : -1.0e30f);
                float bm = redmax16(fmaxf(a, b));
                float mn = fmaxf(m[r], bm);
                float f  = __expf(m[r] - mn);
                m[r] = mn;
                p0[r] = __expf(a - mn);
                p1[r] = __expf(b - mn);
                ls[r]  = ls[r] * f + p0[r] + p1[r];
                acc0[r] *= f;
                acc1[r] *= f;
                int row = r + 8 * hi;
                Pw[row * QS_STR + l15]      = (_Float16)p0[r];
                Pw[row * QS_STR + 16 + l15] = (_Float16)p1[r];
            }
            CFENCE();
            v16h pa  = load_frag_A(Pw + l15 * QS_STR, hi);
            v16h bv0 = load_frag_B(Vs + l15 * VS_STR + kb);
            v16h bv1 = load_frag_B(Vs + (16 + l15) * VS_STR + kb);
            acc0 = wmma_f16(pa, bv0, acc0);
            acc1 = wmma_f16(pa, bv1, acc1);
            CFENCE();
        }

        // finalize + store (token, c) f16
#pragma unroll
        for (int r = 0; r < 8; ++r) {
            float L = redsum16(ls[r]);
            float inv = 1.0f / L;
            size_t tok = (size_t)win * NTOK + qbase + r + 8 * hi;
            attnout[tok * C_DIM + head * DH + l15] = (_Float16)(acc0[r] * inv);
            if (l15 < 8)
                attnout[tok * C_DIM + head * DH + 16 + l15] = (_Float16)(acc1[r] * inv);
        }
    }
}

// ---------------------------------------------------------------------------
// Stage 3: output projection GEMM (98304 x 192) @ (192 x 192)^T + bias (WMMA)
//   grid 768, block 256 (8 waves x 16 rows = 128 tokens per block)
//   LDS: W16[192][200] + Xs[128][200] = 128000 B
// ---------------------------------------------------------------------------
#define PW_STR 200
__global__ void proj_kernel(const _Float16* __restrict__ attnout,
                            const float* __restrict__ proj_w,
                            const float* __restrict__ proj_b,
                            float* __restrict__ projtmp) {
    extern __shared__ char smem[];
    _Float16* W16 = (_Float16*)smem;              // [192][200]  (o, c)
    _Float16* Xs  = W16 + C_DIM * PW_STR;         // [128][200]  (t, c)

    int tid = threadIdx.x;
    int tb = blockIdx.x * 128;

    for (int i = tid; i < C_DIM * C_DIM; i += 256) {
        int o = i / C_DIM, c = i % C_DIM;
        W16[o * PW_STR + c] = (_Float16)proj_w[i];
    }
    for (int i = tid; i < 128 * C_DIM; i += 256) {
        int t = i / C_DIM, c = i % C_DIM;
        Xs[t * PW_STR + c] = attnout[(size_t)(tb + t) * C_DIM + c];
    }
    __syncthreads();

    int wave = tid >> 5, lane = tid & 31;
    int l15 = lane & 15, hi = lane >> 4;
    int rowb = wave * 16;

    v16h af[6];
#pragma unroll
    for (int ch = 0; ch < 6; ++ch)
        af[ch] = load_frag_A(Xs + (rowb + l15) * PW_STR + ch * 32, hi);

    for (int ot = 0; ot < 12; ++ot) {
        v8f acc = {};
#pragma unroll
        for (int ch = 0; ch < 6; ++ch) {
            v16h bf = load_frag_B(W16 + (ot * 16 + l15) * PW_STR + ch * 32);
            acc = wmma_f16(af[ch], bf, acc);
        }
        float bb = proj_b[ot * 16 + l15];
#pragma unroll
        for (int r = 0; r < 8; ++r) {
            size_t tok = (size_t)tb + rowb + r + 8 * hi;
            projtmp[tok * C_DIM + ot * 16 + l15] = acc[r] + bb;
        }
    }
}

// ---------------------------------------------------------------------------
// Stage 4: window reverse + un-shift + transpose -> (C, D, H, W) f32
//   grid (NWIN, 2), block 256; LDS float[192][193] = 148224 B
// ---------------------------------------------------------------------------
__global__ void unpermute_kernel(const float* __restrict__ projtmp,
                                 float* __restrict__ out) {
    extern __shared__ char smem[];
    float* buf = (float*)smem;                    // [192 tokens][193]

    int win = blockIdx.x, half = blockIdx.y;
    int t0 = half * 192;
    int tid = threadIdx.x;

    for (int i = tid; i < 192 * C_DIM; i += 256) {
        int t = i / C_DIM, c = i % C_DIM;
        buf[t * 193 + c] = projtmp[((size_t)win * NTOK + t0 + t) * C_DIM + c];
    }
    __syncthreads();

    int wd = win >> 6, wh = (win >> 3) & 7, ww = win & 7;
    for (int i = tid; i < C_DIM * 192; i += 256) {
        int c = i / 192, tt = i % 192;
        int t = t0 + tt;
        int td = t >> 6, th = (t >> 3) & 7, tw = t & 7;
        int d = (wd * 6 + td + 3) % DD;           // same mapping as load side
        int h = (wh * 8 + th + 4) & 63;
        int w = (ww * 8 + tw + 4) & 63;
        out[(size_t)c * DHW + (size_t)d * (HH * WW_) + h * WW_ + w] =
            buf[tt * 193 + c];
    }
}

// ---------------------------------------------------------------------------
// Host launcher
// ---------------------------------------------------------------------------
extern "C" void kernel_launch(void* const* d_in, const int* in_sizes, int n_in,
                              void* d_out, int out_size, void* d_ws, size_t ws_size,
                              hipStream_t stream) {
    (void)in_sizes; (void)n_in; (void)out_size; (void)ws_size;

    const float* q_map  = (const float*)d_in[0];
    const float* k_map  = (const float*)d_in[1];
    const float* v_map  = (const float*)d_in[2];
    const float* nq_w   = (const float*)d_in[3];
    const float* nq_b   = (const float*)d_in[4];
    const float* nkv_w  = (const float*)d_in[5];
    const float* nkv_b  = (const float*)d_in[6];
    const float* proj_w = (const float*)d_in[7];
    const float* proj_b = (const float*)d_in[8];
    float* outp = (float*)d_out;

    const size_t QKV_BYTES = (size_t)NWIN * C_DIM * NTOK * sizeof(_Float16); // 37.75 MB
    char* ws = (char*)d_ws;
    _Float16* qn      = (_Float16*)(ws);
    _Float16* kn      = (_Float16*)(ws + QKV_BYTES);
    _Float16* vn      = (_Float16*)(ws + 2 * QKV_BYTES);
    _Float16* attnout = (_Float16*)(ws + 3 * QKV_BYTES);
    float*    projtmp = (float*)(ws);  // overlays qn+kn (dead after stage 2)

    const int SMEM1 = C_DIM * NTOK * (int)sizeof(_Float16);                       // 147456
    const int SMEM2 = (2 * NTOK * QS_STR + DHP * VS_STR + 8 * 16 * QS_STR) *
                      (int)sizeof(_Float16);                                      // 85632
    const int SMEM3 = (C_DIM * PW_STR + 128 * PW_STR) * (int)sizeof(_Float16);    // 128000
    const int SMEM4 = 192 * 193 * (int)sizeof(float);                             // 148224

    (void)hipFuncSetAttribute((const void*)ln_window_kernel,
        hipFuncAttributeMaxDynamicSharedMemorySize, SMEM1);
    (void)hipFuncSetAttribute((const void*)attn_kernel,
        hipFuncAttributeMaxDynamicSharedMemorySize, SMEM2);
    (void)hipFuncSetAttribute((const void*)proj_kernel,
        hipFuncAttributeMaxDynamicSharedMemorySize, SMEM3);
    (void)hipFuncSetAttribute((const void*)unpermute_kernel,
        hipFuncAttributeMaxDynamicSharedMemorySize, SMEM4);

    ln_window_kernel<<<dim3(NWIN, 3), 384, SMEM1, stream>>>(
        q_map, k_map, v_map, nq_w, nq_b, nkv_w, nkv_b, qn, kn, vn);

    attn_kernel<<<dim3(NWIN, NH), 256, SMEM2, stream>>>(qn, kn, vn, attnout);

    proj_kernel<<<dim3((NWIN * NTOK) / 128), 256, SMEM3, stream>>>(
        attnout, proj_w, proj_b, projtmp);

    unpermute_kernel<<<dim3(NWIN, 2), 256, SMEM4, stream>>>(projtmp, outp);
}
